// XAIGuidedTransformerLayer_5428838662340
// MI455X (gfx1250) — compile-verified, matrix operand order checked by reading, not run on hardware
//
#include <hip/hip_runtime.h>
#include <hip/hip_bf16.h>

#define BB 8
#define SS 2048
#define DD 512

typedef __attribute__((ext_vector_type(16))) _Float16 v16h;
typedef __attribute__((ext_vector_type(8)))  float    v8f;

union V16H { v16h v; _Float16 h[16]; unsigned u[8]; };
union V8F  { v8f  v; float f[8]; };

__device__ __forceinline__ unsigned pack2(float a, float b) {
    union { _Float16 h[2]; unsigned u; } t;
    t.h[0] = (_Float16)a; t.h[1] = (_Float16)b;
    return t.u;
}

__device__ __forceinline__ v8f wmma_f16(v16h a, v16h b, v8f c) {
    // D = A(16x32 f16) * B(32x16 f16) + C(16x16 f32)
    return __builtin_amdgcn_wmma_f32_16x16x32_f16(false, a, false, b, (short)0, c, false, false);
}

// ---------------------------------------------------------------------------
// Kernel 0a: convert + transpose weights to f16.
// W1T[n][k] = W1[k][n]  (n<1024,k<512);  W2T[n][k] = W2[k][n]  (n,k<512)
// ---------------------------------------------------------------------------
__global__ void prep_weights(const float* __restrict__ W1, const float* __restrict__ W2,
                             _Float16* __restrict__ W1T, _Float16* __restrict__ W2T) {
    int i = blockIdx.x * 256 + threadIdx.x;
    if (i < 1024 * 512) {
        int n = i >> 9, k = i & 511;
        W1T[i] = (_Float16)W1[k * 1024 + n];
    }
    int j = i - 1024 * 512;
    if (j >= 0 && j < 512 * 512) {
        int n = j >> 9, k = j & 511;
        W2T[j] = (_Float16)W2[k * 512 + n];
    }
}

// ---------------------------------------------------------------------------
// Kernel 0b: pre-convert K to f16 (kh[b][s][d]) and pre-transpose X to f16
// (xT[b][d][s]) so all WMMA B-fragments in the attention kernel are single
// aligned dword loads (f16 pairs contiguous along the WMMA K dimension).
// ---------------------------------------------------------------------------
__global__ void prep_kx(const float* __restrict__ k, const float* __restrict__ x,
                        _Float16* __restrict__ kh, _Float16* __restrict__ xT) {
    int i = blockIdx.x * 256 + threadIdx.x;   // i < B*S*D = 8M
    kh[i] = (_Float16)k[i];
    int b = i >> 20;                 // / (S*D)
    int rem = i & ((SS * DD) - 1);
    int s = rem >> 9;                // / D
    int d = rem & (DD - 1);
    xT[((size_t)(b * DD + d) << 11) + s] = (_Float16)x[i];
}

// ---------------------------------------------------------------------------
// Kernel 1: flash attention (single head, head_dim = D = 512) fused with
// residual add + RMSNorm-1.  One block = (batch, 16-query tile), 4 waves.
// Wave w owns d-columns [w*128, w*128+128).  Q fragments hoisted; all
// B-fragments for a chunk preloaded into distinct registers so loads issue
// as one clause ahead of the WMMA burst.
// ---------------------------------------------------------------------------
__global__ void __launch_bounds__(128)
attn_rmsnorm_kernel(const float* __restrict__ x, const float* __restrict__ q,
                    const _Float16* __restrict__ kh, const _Float16* __restrict__ xT,
                    const float* __restrict__ norm1w, float* __restrict__ h_ws) {
    const int b    = blockIdx.x >> 7;          // 128 query-blocks per batch
    const int q0   = (blockIdx.x & 127) << 4;  // 16 queries
    const int tid  = threadIdx.x;
    const int lane = tid & 31;
    const int wave = tid >> 5;
    const int l15  = lane & 15;
    const int lhalf = lane >> 4;

    __shared__ __align__(16) float    Spart[4][16][32];
    __shared__ __align__(16) float    Ssum[16][32];
    __shared__ __align__(16) _Float16 Pl[16][32];
    __shared__ float mstate[16], lstate[16], rowscale[16], ssq[16];

    // 16-bit A/B fragment K-offsets per VGPR index v (ISA 7.12.2 layout)
    int kofs[8];
#pragma unroll
    for (int v = 0; v < 8; ++v)
        kofs[v] = ((v & 4) << 2) + lhalf * 8 + (v & 3) * 2;

    if (tid < 16) { mstate[tid] = -1e30f; lstate[tid] = 0.0f; }
    __syncthreads();

    const int dbaseW = wave * 128;
    const float*    qb  = q  + (size_t)(b * SS + q0) * DD;
    const _Float16* kpb = kh + ((size_t)b * SS) * DD;
    const _Float16* xTb = xT + ((size_t)b * DD) * SS;
    const float*    xb  = x  + (size_t)b * SS * DD;

    // ---- hoisted Q A-fragments (loop-invariant): 4 d-chunks of 32 ---------
    V16H qa[4];
#pragma unroll
    for (int kk = 0; kk < 4; ++kk) {
        const float* qrow = qb + (size_t)l15 * DD + dbaseW + kk * 32;
#pragma unroll
        for (int v = 0; v < 8; ++v)
            qa[kk].u[v] = pack2(qrow[kofs[v]], qrow[kofs[v] + 1]);
    }

    V8F o[8];
#pragma unroll
    for (int t = 0; t < 8; ++t)
#pragma unroll
        for (int r = 0; r < 8; ++r) o[t].f[r] = 0.0f;

    for (int kb = 0; kb < SS; kb += 32) {
        // ---- prefetch next key-block's K and X^T streams ------------------
        if (kb + 32 < SS) {
            __builtin_prefetch(kpb + (size_t)(kb + 32 + l15) * DD + dbaseW, 0, 1);
            __builtin_prefetch(xTb + (((size_t)(dbaseW + l15)) << 11) + kb + 32, 0, 1);
        }

        // ---- partial scores over this wave's 128-wide d slice -------------
        // Preload all 8 K B-fragments (distinct regs) then 8 WMMAs.
        V16H bm[8];
#pragma unroll
        for (int kk = 0; kk < 4; ++kk) {
            const int dchunk = dbaseW + kk * 32;
#pragma unroll
            for (int nt = 0; nt < 2; ++nt) {
                const _Float16* krow = kpb + (size_t)(kb + nt * 16 + l15) * DD + dchunk;
#pragma unroll
                for (int v = 0; v < 8; ++v)
                    bm[kk * 2 + nt].u[v] = *(const unsigned*)&krow[kofs[v]];
            }
        }
        V8F sc[2];
#pragma unroll
        for (int nt = 0; nt < 2; ++nt)
#pragma unroll
            for (int r = 0; r < 8; ++r) sc[nt].f[r] = 0.0f;
#pragma unroll
        for (int kk = 0; kk < 4; ++kk)
#pragma unroll
            for (int nt = 0; nt < 2; ++nt)
                sc[nt].v = wmma_f16(qa[kk].v, bm[kk * 2 + nt].v, sc[nt].v);

#pragma unroll
        for (int nt = 0; nt < 2; ++nt)
#pragma unroll
            for (int r = 0; r < 8; ++r)
                Spart[wave][r + lhalf * 8][nt * 16 + l15] = sc[nt].f[r];
        __syncthreads();

        // ---- reduce the 4 partials (128 threads x 4 entries) --------------
        {
            const int row = tid >> 3, c0 = (tid & 7) * 4;
#pragma unroll
            for (int j = 0; j < 4; ++j) {
                int c = c0 + j;
                float s = Spart[0][row][c] + Spart[1][row][c] +
                          Spart[2][row][c] + Spart[3][row][c];
                Ssum[row][c] = s * 0.044194173824159216f;  // 1/sqrt(512)
            }
        }
        __syncthreads();

        // ---- online softmax (16 row-owner threads) ------------------------
        if (tid < 16) {
            const int row = tid;
            float mold = mstate[row];
            float mx = mold;
#pragma unroll
            for (int c = 0; c < 32; ++c) mx = fmaxf(mx, Ssum[row][c]);
            float corr = __expf(mold - mx);
            float ls = 0.0f;
#pragma unroll
            for (int c = 0; c < 32; ++c) {
                float p = __expf(Ssum[row][c] - mx);
                Pl[row][c] = (_Float16)p;
                ls += p;
            }
            lstate[row] = lstate[row] * corr + ls;
            mstate[row] = mx;
            rowscale[row] = corr;
        }
        __syncthreads();

        // ---- rescale O, accumulate P @ X (B-frags preloaded from xT) ------
        V16H bx[8];
#pragma unroll
        for (int t = 0; t < 8; ++t) {
            const int d = dbaseW + t * 16 + l15;
            const _Float16* xrow = xTb + ((size_t)d << 11) + kb;  // xT[b][d][kb..]
#pragma unroll
            for (int v = 0; v < 8; ++v)
                bx[t].u[v] = *(const unsigned*)&xrow[kofs[v]];
        }
        float rs[8];
#pragma unroll
        for (int r = 0; r < 8; ++r) rs[r] = rowscale[r + lhalf * 8];
#pragma unroll
        for (int t = 0; t < 8; ++t)
#pragma unroll
            for (int r = 0; r < 8; ++r) o[t].f[r] *= rs[r];

        V16H pa;
#pragma unroll
        for (int v = 0; v < 8; ++v)
            pa.u[v] = *(const unsigned*)&Pl[l15][kofs[v]];

#pragma unroll
        for (int t = 0; t < 8; ++t)
            o[t].v = wmma_f16(pa.v, bx[t].v, o[t].v);
        __syncthreads();
    }

    // ---- epilogue: normalize by l, +x residual, RMSNorm, write h ---------
    if (tid < 16) ssq[tid] = 0.0f;
    float linv[8];
#pragma unroll
    for (int r = 0; r < 8; ++r) linv[r] = 1.0f / lstate[r + lhalf * 8];

#pragma unroll
    for (int t = 0; t < 8; ++t) {
        const int d = dbaseW + t * 16 + l15;
#pragma unroll
        for (int r = 0; r < 8; ++r) {
            const int row = r + lhalf * 8;
            float val = o[t].f[r] * linv[r] + xb[(size_t)(q0 + row) * DD + d];
            o[t].f[r] = val;
        }
    }
    __syncthreads();
#pragma unroll
    for (int r = 0; r < 8; ++r) {
        float s = 0.0f;
#pragma unroll
        for (int t = 0; t < 8; ++t) s += o[t].f[r] * o[t].f[r];
        atomicAdd(&ssq[r + lhalf * 8], s);
    }
    __syncthreads();
    float rinv[8];
#pragma unroll
    for (int r = 0; r < 8; ++r)
        rinv[r] = rsqrtf(ssq[r + lhalf * 8] * (1.0f / 512.0f) + 1.1920929e-07f);
#pragma unroll
    for (int t = 0; t < 8; ++t) {
        const int d = dbaseW + t * 16 + l15;
        const float wn = norm1w[d];
#pragma unroll
        for (int r = 0; r < 8; ++r) {
            const int row = r + lhalf * 8;
            h_ws[(size_t)(b * SS + q0 + row) * DD + d] = o[t].f[r] * rinv[r] * wn;
        }
    }
}

// ---------------------------------------------------------------------------
// Kernel 2: GeGLU FFN fused with residual + RMSNorm-2.
// One block = 16 rows of [B*S, D], 4 waves; wave w owns 128 proj columns of
// both x1 and x2 (so GeGLU is elementwise on matching accumulator tiles).
// B-fragments preloaded in groups of 4 to overlap loads with WMMA.
// ---------------------------------------------------------------------------
__global__ void __launch_bounds__(128)
ffn_rmsnorm_kernel(const float* __restrict__ h, const _Float16* __restrict__ W1T,
                   const float* __restrict__ b1, const _Float16* __restrict__ W2T,
                   const float* __restrict__ b2, const float* __restrict__ norm2w,
                   float* __restrict__ out) {
    const int row0 = blockIdx.x << 4;
    const int tid  = threadIdx.x;
    const int lane = tid & 31;
    const int wave = tid >> 5;
    const int l15  = lane & 15;
    const int lhalf = lane >> 4;
    const int HP = 520;  // padded LDS row stride (halves)

    __shared__ __align__(16) _Float16 hs[16 * 520];
    __shared__ __align__(16) _Float16 gs[16 * 520];
    __shared__ float ssq[16];

    int kofs[8];
#pragma unroll
    for (int v = 0; v < 8; ++v)
        kofs[v] = ((v & 4) << 2) + lhalf * 8 + (v & 3) * 2;

    for (int i = tid; i < 16 * 512; i += 128) {
        int r = i >> 9, c = i & 511;
        hs[r * HP + c] = (_Float16)h[(size_t)(row0 + r) * DD + c];
    }
    __syncthreads();

    // ---- proj = h @ W1 : 16 tiles (8 x1-cols, 8 x2-cols) ------------------
    V8F acc[16];
#pragma unroll
    for (int t = 0; t < 16; ++t)
#pragma unroll
        for (int r = 0; r < 8; ++r) acc[t].f[r] = 0.0f;

    const int nbase = wave * 128;
    for (int kk = 0; kk < 16; ++kk) {
        const int kchunk = kk * 32;
        V16H a;
#pragma unroll
        for (int v = 0; v < 8; ++v)
            a.u[v] = *(const unsigned*)&hs[l15 * HP + kchunk + kofs[v]];
#pragma unroll
        for (int g = 0; g < 4; ++g) {
            V16H bm[4];
#pragma unroll
            for (int j = 0; j < 4; ++j) {
                const int t = g * 4 + j;
                const int n = (t < 8) ? (nbase + t * 16 + l15)
                                      : (512 + nbase + (t - 8) * 16 + l15);
#pragma unroll
                for (int v = 0; v < 8; ++v)
                    bm[j].u[v] = *(const unsigned*)&W1T[(size_t)n * 512 + kchunk + kofs[v]];
            }
#pragma unroll
            for (int j = 0; j < 4; ++j)
                acc[g * 4 + j].v = wmma_f16(a.v, bm[j].v, acc[g * 4 + j].v);
        }
    }

    // ---- GeGLU: g = gelu_exact(x1 + b1a) * (x2 + b1b), write g to LDS -----
#pragma unroll
    for (int j = 0; j < 8; ++j) {
        const int n1 = nbase + j * 16 + l15;
        const float ba = b1[n1], bb = b1[512 + n1];
#pragma unroll
        for (int r = 0; r < 8; ++r) {
            float u  = acc[j].f[r] + ba;
            float x2 = acc[j + 8].f[r] + bb;
            float g  = 0.5f * u * (1.0f + erff(u * 0.70710678118654752f)) * x2;
            gs[(r + lhalf * 8) * HP + n1] = (_Float16)g;
        }
    }
    __syncthreads();

    // ---- ff = g @ W2 ------------------------------------------------------
    V8F oacc[8];
#pragma unroll
    for (int t = 0; t < 8; ++t)
#pragma unroll
        for (int r = 0; r < 8; ++r) oacc[t].f[r] = 0.0f;

    for (int kk = 0; kk < 16; ++kk) {
        const int kchunk = kk * 32;
        V16H a;
#pragma unroll
        for (int v = 0; v < 8; ++v)
            a.u[v] = *(const unsigned*)&gs[l15 * HP + kchunk + kofs[v]];
#pragma unroll
        for (int g = 0; g < 2; ++g) {
            V16H bm[4];
#pragma unroll
            for (int j = 0; j < 4; ++j) {
                const int n = nbase + (g * 4 + j) * 16 + l15;
#pragma unroll
                for (int v = 0; v < 8; ++v)
                    bm[j].u[v] = *(const unsigned*)&W2T[(size_t)n * 512 + kchunk + kofs[v]];
            }
#pragma unroll
            for (int j = 0; j < 4; ++j)
                oacc[g * 4 + j].v = wmma_f16(a.v, bm[j].v, oacc[g * 4 + j].v);
        }
    }

    // ---- epilogue: +b2, +h residual, RMSNorm, store -----------------------
    if (tid < 16) ssq[tid] = 0.0f;
#pragma unroll
    for (int t = 0; t < 8; ++t) {
        const int d = nbase + t * 16 + l15;
        const float bv = b2[d];
#pragma unroll
        for (int r = 0; r < 8; ++r) {
            const int row = r + lhalf * 8;
            float val = oacc[t].f[r] + bv + h[(size_t)(row0 + row) * DD + d];
            oacc[t].f[r] = val;
        }
    }
    __syncthreads();
#pragma unroll
    for (int r = 0; r < 8; ++r) {
        float s = 0.0f;
#pragma unroll
        for (int t = 0; t < 8; ++t) s += oacc[t].f[r] * oacc[t].f[r];
        atomicAdd(&ssq[r + lhalf * 8], s);
    }
    __syncthreads();
    float rinv[8];
#pragma unroll
    for (int r = 0; r < 8; ++r)
        rinv[r] = rsqrtf(ssq[r + lhalf * 8] * (1.0f / 512.0f) + 1.1920929e-07f);
#pragma unroll
    for (int t = 0; t < 8; ++t) {
        const int d = nbase + t * 16 + l15;
        const float wn = norm2w[d];
#pragma unroll
        for (int r = 0; r < 8; ++r) {
            const int row = r + lhalf * 8;
            out[(size_t)(row0 + row) * DD + d] = oacc[t].f[r] * rinv[r] * wn;
        }
    }
}

// ---------------------------------------------------------------------------
extern "C" void kernel_launch(void* const* d_in, const int* in_sizes, int n_in,
                              void* d_out, int out_size, void* d_ws, size_t ws_size,
                              hipStream_t stream) {
    const float* x   = (const float*)d_in[0];
    const float* q   = (const float*)d_in[1];
    const float* k   = (const float*)d_in[2];
    const float* W1  = (const float*)d_in[3];
    const float* b1  = (const float*)d_in[4];
    const float* W2  = (const float*)d_in[5];
    const float* b2  = (const float*)d_in[6];
    const float* n1w = (const float*)d_in[7];
    const float* n2w = (const float*)d_in[8];
    float* out = (float*)d_out;

    // workspace layout:
    //   h   f32  [B,S,D]      32 MB
    //   W1T f16  [1024,512]    1 MB
    //   W2T f16  [512,512]   0.5 MB
    //   kh  f16  [B,S,D]      16 MB
    //   xT  f16  [B,D,S]      16 MB
    char* ws = (char*)d_ws;
    float*    h_ws = (float*)ws;
    _Float16* W1T  = (_Float16*)(ws + (size_t)BB * SS * DD * 4);
    _Float16* W2T  = W1T + 1024 * 512;
    _Float16* kh   = W2T + 512 * 512;
    _Float16* xT   = kh + (size_t)BB * SS * DD;

    prep_weights<<<3072, 256, 0, stream>>>(W1, W2, W1T, W2T);
    prep_kx<<<(BB * SS * DD) / 256, 256, 0, stream>>>(k, x, kh, xT);
    attn_rmsnorm_kernel<<<BB * (SS / 16), 128, 0, stream>>>(x, q, kh, xT, n1w, h_ws);
    ffn_rmsnorm_kernel<<<(BB * SS) / 16, 128, 0, stream>>>(h_ws, W1T, b1, W2T, b2, n2w, out);
}
